// PAB_31009663877215
// MI455X (gfx1250) — compile-verified, matrix operand order checked by reading, not run on hardware
//
#include <hip/hip_runtime.h>
#include <hip/hip_bf16.h>

typedef __attribute__((ext_vector_type(16))) __bf16 v16bf;
typedef __attribute__((ext_vector_type(8)))  float  v8f;
typedef int v4i __attribute__((vector_size(16)));

#define CIN   512
#define HWSZ  4096
#define IMGW  64
#define PABN  64

__device__ __forceinline__ unsigned short f2bf(float f) {
  unsigned u = __float_as_uint(f);
  u += 0x7FFFu + ((u >> 16) & 1u);          // round-to-nearest-even
  return (unsigned short)(u >> 16);
}
__device__ __forceinline__ float bf2f(unsigned short h) {
  return __uint_as_float(((unsigned)h) << 16);
}

// ---------------------------------------------------------------------------
// CDNA5 WMMA fragment order (ISA 7.12.2), used both for LDS tiles and for the
// "fragment-tile" global layouts: a 16(row/col) x 32(K) tile is 512 bf16 =
// 1KB, laid out [lane 0..31][elem 0..15]; lane = ((k&31)>>3 &1)<<4 | (rc&15),
// elem = (k&7) + ((k>>4)&1)*8. Tiles are ordered k-chunk-major then rc-group,
// so a 128-wide x 32-deep operand block is 8 consecutive tiles = 8KB
// contiguous -> stageable with plain async 16B copies.
__device__ __forceinline__ size_t frag_idx(int rc, int k, int ntiles16) {
  const int kl = k & 31;
  const int ln = (((kl >> 3) & 1) << 4) | (rc & 15);
  const int j  = (kl & 7) + ((kl >> 4) << 3);
  return (((size_t)(k >> 5) * ntiles16 + (rc >> 4)) << 9) + (ln << 4) + j;
}

// ---- async global->LDS 16B copy (GLOBAL_LOAD_ASYNC_TO_LDS_B128) -----------
__device__ __forceinline__ void async_ld16(void* l, const void* g) {
#if __has_builtin(__builtin_amdgcn_global_load_async_to_lds_b128)
  __builtin_amdgcn_global_load_async_to_lds_b128((v4i*)g, (v4i*)l, 0, 0);
#else
  const unsigned loff = (unsigned)(size_t)l;   // flat LDS addr[31:0] == LDS offset
  asm volatile("global_load_async_to_lds_b128 %0, %1, off"
               :: "v"(loff), "v"(g) : "memory");
#endif
}
__device__ __forceinline__ void wait_async0() {
#if __has_builtin(__builtin_amdgcn_s_wait_asynccnt)
  __builtin_amdgcn_s_wait_asynccnt(0);
#else
  asm volatile("s_wait_asynccnt 0" ::: "memory");
#endif
}
__device__ __forceinline__ void wait_async2() {   // allow next tile-pair in flight
#if __has_builtin(__builtin_amdgcn_s_wait_asynccnt)
  __builtin_amdgcn_s_wait_asynccnt(2);
#else
  asm volatile("s_wait_asynccnt 2" ::: "memory");
#endif
}

// ---- VALU staging (f32 sources needing conversion) ------------------------
template <typename LoaderA, typename LoaderB>
__device__ __forceinline__ void stage_tiles64(unsigned short (*la)[32][16],
                                              unsigned short (*lb)[32][16],
                                              int t, LoaderA ldA, LoaderB ldB) {
  const int st = t >> 7;
  const int ln = (t >> 2) & 31;
  const int jb = (t & 3) * 4;
  const int rc = st * 16 + (ln & 15);
  const int kb = (ln >> 4) << 3;
#pragma unroll
  for (int jj = 0; jj < 4; ++jj) {
    const int j = jb + jj;
    const int k = kb + (j & 7) + ((j & 8) << 1);
    la[st][ln][j] = ldA(rc, k);
    lb[st][ln][j] = ldB(k, rc);
  }
}

template <typename LoaderA, typename LoaderB>
__device__ __forceinline__ void stage_tiles128(unsigned short (*la)[32][16],
                                               unsigned short (*lb)[32][16],
                                               int t, LoaderA ldA, LoaderB ldB) {
  const int ln = (t >> 2) & 31;
  const int jb = (t & 3) * 4;
  const int kb = (ln >> 4) << 3;
#pragma unroll
  for (int g = 0; g < 2; ++g) {
    const int st = (t >> 7) + 4 * g;
    const int rc = st * 16 + (ln & 15);
#pragma unroll
    for (int jj = 0; jj < 4; ++jj) {
      const int j = jb + jj;
      const int k = kb + (j & 7) + ((j & 8) << 1);
      la[st][ln][j] = ldA(rc, k);
      lb[st][ln][j] = ldB(k, rc);
    }
  }
}

__device__ __forceinline__ v16bf frag(const unsigned short (*l)[32][16],
                                      int st, int lane) {
  return *reinterpret_cast<const v16bf*>(&l[st][lane][0]);
}

__device__ __forceinline__ v8f wmma_bf16(v16bf a, v16bf b, v8f c) {
  return __builtin_amdgcn_wmma_f32_16x16x32_bf16(false, a, false, b,
                                                 (short)0, c, false, false);
}

// ---------------- 1x1 conv: D[64,4096] = W[64,512] @ X[512,4096] + bias -----
// Output stored in fragment-tile order keyed to how k_logits consumes it:
// both xtop (logits B) and xcen (logits A) map (position->rc, channel->k).
__global__ __launch_bounds__(512)
void k_gemm1x1(const float* __restrict__ w, const float* __restrict__ bias,
               const float* __restrict__ x, unsigned short* __restrict__ out) {
  __shared__ __align__(32) unsigned short la[4][32][16];
  __shared__ __align__(32) unsigned short lb[4][32][16];
  const int t = threadIdx.x;
  const int n0 = blockIdx.x * 64;
  const int bidx = blockIdx.y;
  const float* xb = x + (size_t)bidx * CIN * HWSZ;
  const int lane = t & 31, wv = t >> 5, mt = wv >> 2, nt = wv & 3;
  v8f acc = {};
  for (int k0 = 0; k0 < CIN; k0 += 32) {
    __syncthreads();
    stage_tiles64(la, lb, t,
      [&](int m, int k) { return f2bf(w[(size_t)m * CIN + k0 + k]); },
      [&](int k, int n) { return f2bf(xb[(size_t)(k0 + k) * HWSZ + n0 + n]); });
    __syncthreads();
    acc = wmma_bf16(frag(la, mt, lane), frag(lb, nt, lane), acc);
  }
  const int hi = (lane >> 4) << 3;
#pragma unroll
  for (int r = 0; r < 8; ++r) {
    const int m = mt * 16 + hi + r;                 // pab channel = GEMM k later
    const int n = n0 + nt * 16 + (lane & 15);       // position    = GEMM rc later
    const float val = acc[r] + bias[m];
    out[(size_t)bidx * PABN * HWSZ + frag_idx(n, m, HWSZ / 16)] = f2bf(val);
  }
}

// ---------------- 3x3 conv as 9 shifted K=512 GEMMs, 128x128 blocks ---------
// mode 0: fragment-tile bf16 V (PV's B: k=position, rc=channel)
// mode 1: f32 out[b][o][p]
__global__ __launch_bounds__(512)
void k_conv3(const float* __restrict__ xin, const float* __restrict__ wgt,
             const float* __restrict__ bias, unsigned short* __restrict__ out_bf,
             float* __restrict__ out_f32, int mode) {
  __shared__ __align__(32) unsigned short la[8][32][16];
  __shared__ __align__(32) unsigned short lb[8][32][16];
  const int t = threadIdx.x;
  const int p0 = blockIdx.x * 128;   // 128 positions = two image rows
  const int h0 = blockIdx.x * 2;
  const int m0 = blockIdx.y * 128;   // out-channel tile
  const int bidx = blockIdx.z;
  const float* xb = xin + (size_t)bidx * CIN * HWSZ;
  const int lane = t & 31, wv = t >> 5, wm = wv >> 2, wn = wv & 3;
  v8f acc[2][2] = {};
  for (int tap = 0; tap < 9; ++tap) {
    const int kh = tap / 3 - 1, kw = tap % 3 - 1;
    for (int k0 = 0; k0 < CIN; k0 += 32) {
      __syncthreads();
      stage_tiles128(la, lb, t,
        [&](int m, int k) {
          return f2bf(wgt[((size_t)(m0 + m) * CIN + (k0 + k)) * 9 + tap]);
        },
        [&](int k, int n) {
          const int hh = h0 + (n >> 6) + kh;
          const int wc = (n & 63) + kw;
          const float f = (hh >= 0 && hh < IMGW && wc >= 0 && wc < IMGW)
                              ? xb[(size_t)(k0 + k) * HWSZ + hh * IMGW + wc]
                              : 0.f;
          return f2bf(f);
        });
      __syncthreads();
      const v16bf a0 = frag(la, wm * 2 + 0, lane);
      const v16bf a1 = frag(la, wm * 2 + 1, lane);
      const v16bf b0 = frag(lb, wn * 2 + 0, lane);
      const v16bf b1 = frag(lb, wn * 2 + 1, lane);
      acc[0][0] = wmma_bf16(a0, b0, acc[0][0]);
      acc[0][1] = wmma_bf16(a0, b1, acc[0][1]);
      acc[1][0] = wmma_bf16(a1, b0, acc[1][0]);
      acc[1][1] = wmma_bf16(a1, b1, acc[1][1]);
    }
  }
  const int hi = (lane >> 4) << 3;
#pragma unroll
  for (int i = 0; i < 2; ++i)
#pragma unroll
    for (int j = 0; j < 2; ++j)
#pragma unroll
      for (int r = 0; r < 8; ++r) {
        const int m = m0 + (wm * 2 + i) * 16 + hi + r;
        const int p = p0 + (wn * 2 + j) * 16 + (lane & 15);
        const float val = acc[i][j][r] + bias[m];
        if (mode == 0)
          out_bf[(size_t)bidx * HWSZ * CIN + frag_idx(m, p, CIN / 16)] = f2bf(val);
        else
          out_f32[(size_t)bidx * CIN * HWSZ + (size_t)m * HWSZ + p] = val;
      }
}

// ---------------- logits: S = Xc^T[4096,64] @ Xt[64,4096], + block max ------
// K = 64 = two tile-pairs: prefetch both via async, one barrier, 8 WMMAs.
__global__ __launch_bounds__(512)
void k_logits(const unsigned short* __restrict__ xcen,
              const unsigned short* __restrict__ xtop,
              unsigned short* __restrict__ S, float* __restrict__ pmax) {
  __shared__ __align__(32) unsigned short la[2][8][32][16];
  __shared__ __align__(32) unsigned short lb[2][8][32][16];
  __shared__ float red[512];
  const int t = threadIdx.x;
  const int n0 = blockIdx.x * 128;   // S row tile
  const int c0 = blockIdx.y * 128;   // S col tile
  const int lane = t & 31, wv = t >> 5, wm = wv >> 2, wn = wv & 3;
#pragma unroll
  for (int kc = 0; kc < 2; ++kc) {
    const size_t atile = ((size_t)(kc * (HWSZ / 16) + (n0 >> 4)) << 9);
    const size_t btile = ((size_t)(kc * (HWSZ / 16) + (c0 >> 4)) << 9);
    async_ld16((char*)la[kc] + t * 16, (const char*)(xcen + atile) + t * 16);
    async_ld16((char*)lb[kc] + t * 16, (const char*)(xtop + btile) + t * 16);
  }
  wait_async0();
  __syncthreads();
  v8f acc[2][2] = {};
#pragma unroll
  for (int kc = 0; kc < 2; ++kc) {
    const v16bf a0 = frag(la[kc], wm * 2 + 0, lane);
    const v16bf a1 = frag(la[kc], wm * 2 + 1, lane);
    const v16bf b0 = frag(lb[kc], wn * 2 + 0, lane);
    const v16bf b1 = frag(lb[kc], wn * 2 + 1, lane);
    acc[0][0] = wmma_bf16(a0, b0, acc[0][0]);
    acc[0][1] = wmma_bf16(a0, b1, acc[0][1]);
    acc[1][0] = wmma_bf16(a1, b0, acc[1][0]);
    acc[1][1] = wmma_bf16(a1, b1, acc[1][1]);
  }
  const int hi = (lane >> 4) << 3;
  float lm = -3.4e38f;
#pragma unroll
  for (int i = 0; i < 2; ++i)
#pragma unroll
    for (int j = 0; j < 2; ++j)
#pragma unroll
      for (int r = 0; r < 8; ++r) {
        const int row = n0 + (wm * 2 + i) * 16 + hi + r;
        const int col = c0 + (wn * 2 + j) * 16 + (lane & 15);
        S[(size_t)row * HWSZ + col] = f2bf(acc[i][j][r]);
        lm = fmaxf(lm, acc[i][j][r]);
      }
  red[t] = lm;
  __syncthreads();
  for (int s = 256; s > 0; s >>= 1) {
    if (t < s) red[t] = fmaxf(red[t], red[t + s]);
    __syncthreads();
  }
  if (t == 0) pmax[blockIdx.y * gridDim.x + blockIdx.x] = red[0];
}

// ---------------- sum of exp(S - max); also materialize P = exp(S - max) ----
// P written in fragment-tile order (rc = S row, k = S col) so k_pv's A operand
// is a pure async copy. exp is computed exactly once per logit.
__global__ __launch_bounds__(256)
void k_sumexp(const unsigned short* __restrict__ S, const float* __restrict__ gmax,
              unsigned short* __restrict__ P, float* __restrict__ psum) {
  __shared__ float red[256];
  const int t = threadIdx.x;
  const float mx = gmax[0];
  const size_t total = (size_t)HWSZ * HWSZ;
  const size_t stride = (size_t)gridDim.x * 256;
  float s = 0.f;
  for (size_t i = (size_t)blockIdx.x * 256 + t; i < total; i += stride) {
    const float e = __expf(bf2f(S[i]) - mx);
    s += e;
    const int row = (int)(i >> 12);          // HWSZ = 4096
    const int col = (int)(i & (HWSZ - 1));
    P[frag_idx(row, col, HWSZ / 16)] = f2bf(e);
  }
  red[t] = s;
  __syncthreads();
  for (int k = 128; k > 0; k >>= 1) {
    if (t < k) red[t] += red[t + k];
    __syncthreads();
  }
  if (t == 0) psum[blockIdx.x] = red[0];
}

// ---------------- deterministic final reduction (op 0 = max, 1 = sum) -------
__global__ __launch_bounds__(256)
void k_reduce(const float* __restrict__ in, int n, float* __restrict__ out, int op) {
  __shared__ float red[256];
  const int t = threadIdx.x;
  float v = op ? 0.f : -3.4e38f;
  for (int i = t; i < n; i += 256) v = op ? (v + in[i]) : fmaxf(v, in[i]);
  red[t] = v;
  __syncthreads();
  for (int s = 128; s > 0; s >>= 1) {
    if (t < s) red[t] = op ? (red[t] + red[t + s]) : fmaxf(red[t], red[t + s]);
    __syncthreads();
  }
  if (t == 0) *out = red[0];
}

// ---------------- P@V: both operands async fragment tiles, double-buffered --
// Per iteration: issue async pair for tile i+1, s_wait_asynccnt 2 (exactly the
// next pair outstanding, ASYNCcnt is in-order) -> tile i resident; barrier;
// 4 WMMAs; barrier (readers done before buffer reuse).
__global__ __launch_bounds__(512)
void k_pv(const unsigned short* __restrict__ Pm, const unsigned short* __restrict__ v,
          const float* __restrict__ xb, const float* __restrict__ gsump,
          float* __restrict__ zb) {
  __shared__ __align__(32) unsigned short la[2][8][32][16];
  __shared__ __align__(32) unsigned short lb[2][8][32][16];
  const int t = threadIdx.x;
  const int n0 = blockIdx.x * 128;   // output positions
  const int c0 = blockIdx.y * 128;   // output channels
  const float inv = 1.0f / gsump[0];
  const int lane = t & 31, wv = t >> 5, wm = wv >> 2, wn = wv & 3;
  auto issue = [&](int i) {
    const size_t atile = ((size_t)(i * (HWSZ / 16) + (n0 >> 4)) << 9);
    const size_t btile = ((size_t)(i * (CIN / 16) + (c0 >> 4)) << 9);
    async_ld16((char*)la[i & 1] + t * 16, (const char*)(Pm + atile) + t * 16);
    async_ld16((char*)lb[i & 1] + t * 16, (const char*)(v + btile) + t * 16);
  };
  v8f acc[2][2] = {};
  const int NIT = HWSZ / 32;   // 128
  issue(0);
  for (int i = 0; i < NIT; ++i) {
    if (i + 1 < NIT) {
      issue(i + 1);
      wait_async2();
    } else {
      wait_async0();
    }
    __syncthreads();
    const int cur = i & 1;
    const v16bf a0 = frag(la[cur], wm * 2 + 0, lane);
    const v16bf a1 = frag(la[cur], wm * 2 + 1, lane);
    const v16bf b0 = frag(lb[cur], wn * 2 + 0, lane);
    const v16bf b1 = frag(lb[cur], wn * 2 + 1, lane);
    acc[0][0] = wmma_bf16(a0, b0, acc[0][0]);
    acc[0][1] = wmma_bf16(a0, b1, acc[0][1]);
    acc[1][0] = wmma_bf16(a1, b0, acc[1][0]);
    acc[1][1] = wmma_bf16(a1, b1, acc[1][1]);
    __syncthreads();
  }
  const int hi = (lane >> 4) << 3;
#pragma unroll
  for (int i = 0; i < 2; ++i)
#pragma unroll
    for (int j = 0; j < 2; ++j)
#pragma unroll
      for (int r = 0; r < 8; ++r) {
        const int row = n0 + (wm * 2 + i) * 16 + hi + r;
        const int col = c0 + (wn * 2 + j) * 16 + (lane & 15);
        const size_t idx = (size_t)row * CIN + col;
        // raw reshape: attn flat index space == x flat index space per batch
        zb[idx] = xb[idx] + acc[i][j][r] * inv;
      }
}

extern "C" void kernel_launch(void* const* d_in, const int* in_sizes, int n_in,
                              void* d_out, int out_size, void* d_ws, size_t ws_size,
                              hipStream_t stream) {
  (void)in_sizes; (void)n_in; (void)out_size; (void)ws_size;
  const float* x        = (const float*)d_in[0];
  const float* w_top    = (const float*)d_in[1];
  const float* b_top    = (const float*)d_in[2];
  const float* w_center = (const float*)d_in[3];
  const float* b_center = (const float*)d_in[4];
  const float* w_bottom = (const float*)d_in[5];
  const float* b_bottom = (const float*)d_in[6];
  const float* w_out    = (const float*)d_in[7];
  const float* b_out    = (const float*)d_in[8];

  char* p = (char*)d_ws;
  auto carve = [&](size_t bytes) {
    char* r = p;
    p += (bytes + 255) & ~(size_t)255;
    return r;
  };
  unsigned short* xtop = (unsigned short*)carve((size_t)4 * PABN * HWSZ * 2); // frag tiles
  unsigned short* xcen = (unsigned short*)carve((size_t)4 * HWSZ * PABN * 2); // frag tiles
  unsigned short* vbuf = (unsigned short*)carve((size_t)4 * HWSZ * CIN * 2);  // frag tiles
  unsigned short* Sbuf = (unsigned short*)carve((size_t)HWSZ * HWSZ * 2);     // per-batch reuse
  unsigned short* Pbuf = (unsigned short*)carve((size_t)HWSZ * HWSZ * 2);     // frag tiles
  float* zbuf = (float*)carve((size_t)4 * CIN * HWSZ * 4);                    // x + sp_map
  float* pmax = (float*)carve(4096 * sizeof(float));
  float* psum = (float*)carve(2048 * sizeof(float));
  float* gmax = (float*)carve(4 * sizeof(float));
  float* gsum = (float*)carve(4 * sizeof(float));

  k_gemm1x1<<<dim3(64, 4), 512, 0, stream>>>(w_top, b_top, x, xtop);
  k_gemm1x1<<<dim3(64, 4), 512, 0, stream>>>(w_center, b_center, x, xcen);
  k_conv3<<<dim3(32, 4, 4), 512, 0, stream>>>(x, w_bottom, b_bottom, vbuf, nullptr, 0);

  for (int b = 0; b < 4; ++b) {
    k_logits<<<dim3(32, 32), 512, 0, stream>>>(xcen + (size_t)b * HWSZ * PABN,
                                               xtop + (size_t)b * PABN * HWSZ,
                                               Sbuf, pmax);
    k_reduce<<<1, 256, 0, stream>>>(pmax, 1024, gmax + b, 0);
    k_sumexp<<<2048, 256, 0, stream>>>(Sbuf, gmax + b, Pbuf, psum);
    k_reduce<<<1, 256, 0, stream>>>(psum, 2048, gsum + b, 1);
    k_pv<<<dim3(32, 4), 512, 0, stream>>>(Pbuf, vbuf + (size_t)b * HWSZ * CIN,
                                          x + (size_t)b * CIN * HWSZ,
                                          gsum + b,
                                          zbuf + (size_t)b * CIN * HWSZ);
  }

  k_conv3<<<dim3(32, 4, 4), 512, 0, stream>>>(zbuf, w_out, b_out, nullptr,
                                              (float*)d_out, 1);
}
